// Attention_39676907885369
// MI455X (gfx1250) — compile-verified, matrix operand order checked by reading, not run on hardware
//
#include <hip/hip_runtime.h>
#include <hip/hip_bf16.h>

// ---------------------------------------------------------------------------
// MI455X (gfx1250, wave32) fused projection + attention.
//
// Roofline: ~95 GFLOP of matmul vs >= 1 GB of mandatory HBM traffic
// (384 MB in, 640 MB out) -> ~44 us memory floor @ 23.3 TB/s. All matmuls run
// on the f16 WMMA path (v_wmma_f32_16x16x32_f16, f32 accumulate) so compute
// sits far below the memory roof; the softmax multi-pass over the attention
// buffer stays in the 192 MB L2 (256 KB working set per workgroup).
//
// f16 tile staging (Q, K, V^T -> LDS) uses GLOBAL_LOAD_ASYNC_TO_LDS_B128
// (ASYNCcnt) via inline asm so the DMA overlaps the exp/score VALU work and
// no VGPRs are burned on copies. V is pre-transposed per head by the
// projection kernel so the attention kernel never transposes anything.
// ---------------------------------------------------------------------------

typedef __attribute__((ext_vector_type(16))) _Float16 v16h;
typedef __attribute__((ext_vector_type(8)))  _Float16 v8h;
typedef __attribute__((ext_vector_type(8)))  float    v8f;

#define LDS_STRIDE 136   // 128 halves + 8 pad halves -> breaks LDS bank conflicts
#define SEQ   512
#define DIM   128
#define NHEAD 512

// Async DMA: global -> LDS, 16 bytes per lane. LDS byte address is the low
// 32 bits of the generic pointer (ISA: LDS aperture maps addr[31:0]).
__device__ __forceinline__ void async_copy_b128(const void* gbase, unsigned goff,
                                                void* ldst) {
  asm volatile("global_load_async_to_lds_b128 %0, %1, %2"
               :
               : "v"((unsigned)(unsigned long long)ldst),
                 "v"(goff),
                 "s"((unsigned long long)gbase)
               : "memory");
}
__device__ __forceinline__ void wait_async0() {
  asm volatile("s_wait_asynccnt 0" ::: "memory");
}

// Load a 16x32 f16 WMMA fragment from a row-major padded LDS tile.
// 16-bit A layout: lanes 0-15 hold rows 0-15 with K=[0..8)+[16..24),
// lanes 16-31 the same rows with K=[8..16)+[24..32): two contiguous 16-byte
// chunks -> ds_load_b128. B is staged as B^T row-major and uses the same load.
__device__ __forceinline__ v16h load_frag_row(const _Float16* __restrict__ base,
                                              int rowBase, int tk) {
  const int l  = threadIdx.x & 31;
  const int hi = l >> 4;
  const _Float16* p = base + (rowBase + (l & 15)) * LDS_STRIDE + tk + 8 * hi;
  v8h lo  = *(const v8h*)p;
  v8h hic = *(const v8h*)(p + 16);
  v16h f;
#pragma unroll
  for (int i = 0; i < 8; ++i) { f[i] = lo[i]; f[i + 8] = hic[i]; }
  return f;
}

// ---------------------------------------------------------------------------
// Kernel 1: projection. The reference reshape (a,b,b,c)->(n,c,h)->transpose
// reduces to:  P[n, 128*hb + k, c] = X[n, 4c+hb, :] . W[k, :] + b[k]
// -> per (tensor, n, hb) a 128x128x128 GEMM.
// Q, K stored f16 head-major (n, h, c); V stored f16 head-TRANSPOSED
// (n, c, h) so the attention kernel can async-copy B^T tiles directly.
// ---------------------------------------------------------------------------
__global__ __launch_bounds__(256) void proj_kernel(
    const float* __restrict__ q_in, const float* __restrict__ k_in,
    const float* __restrict__ v_in,
    const float* __restrict__ Wq, const float* __restrict__ bq,
    const float* __restrict__ Wk, const float* __restrict__ bk,
    const float* __restrict__ Wv, const float* __restrict__ bv,
    _Float16* __restrict__ qp, _Float16* __restrict__ kp,
    _Float16* __restrict__ vt)
{
  __shared__ _Float16 Asub[128 * LDS_STRIDE];  // X rows (M=c, K=ch), row-major
  __shared__ _Float16 Wt  [128 * LDS_STRIDE];  // W rows = B^T (N=k, K=ch)
  __shared__ float    bs[128];

  const int blk    = blockIdx.x;
  const int tensor = blk >> 11;          // 2048 blocks per tensor
  const int n      = (blk >> 2) & 511;
  const int hb     = blk & 3;

  const float* x; const float* W; const float* b; _Float16* dst;
  if (tensor == 0)      { x = q_in; W = Wq; b = bq; dst = qp; }
  else if (tensor == 1) { x = k_in; W = Wk; b = bk; dst = kp; }
  else                  { x = v_in; W = Wv; b = bv; dst = vt; }

  const int t    = threadIdx.x;
  const int srow = t >> 1;               // 2 threads per row
  const int scol = (t & 1) * 64;

  if (t < 128) bs[t] = b[t];

  // Stage X_sub row (c = srow -> global row j = 4c+hb) and W row, f32 -> f16.
  {
    const float* xs = x + (size_t)n * (SEQ * DIM) + (size_t)(4 * srow + hb) * DIM + scol;
    const float* ws = W + (size_t)srow * DIM + scol;
    _Float16* ad = Asub + srow * LDS_STRIDE + scol;
    _Float16* wd = Wt   + srow * LDS_STRIDE + scol;
#pragma unroll
    for (int i = 0; i < 64; i += 8) {
      float4 f0 = *(const float4*)(xs + i);
      float4 f1 = *(const float4*)(xs + i + 4);
      v8h h;
      h[0]=(_Float16)f0.x; h[1]=(_Float16)f0.y; h[2]=(_Float16)f0.z; h[3]=(_Float16)f0.w;
      h[4]=(_Float16)f1.x; h[5]=(_Float16)f1.y; h[6]=(_Float16)f1.z; h[7]=(_Float16)f1.w;
      *(v8h*)(ad + i) = h;
      float4 g0 = *(const float4*)(ws + i);
      float4 g1 = *(const float4*)(ws + i + 4);
      v8h hw;
      hw[0]=(_Float16)g0.x; hw[1]=(_Float16)g0.y; hw[2]=(_Float16)g0.z; hw[3]=(_Float16)g0.w;
      hw[4]=(_Float16)g1.x; hw[5]=(_Float16)g1.y; hw[6]=(_Float16)g1.z; hw[7]=(_Float16)g1.w;
      *(v8h*)(wd + i) = hw;
    }
  }
  __syncthreads();

  const int w  = t >> 5;                 // wave 0..7, owns M rows [16w, 16w+16)
  const int l  = t & 31;
  const int hi = l >> 4;

#pragma unroll
  for (int tn = 0; tn < 8; ++tn) {       // N (output channel k) tiles
    v8f acc = {};
#pragma unroll
    for (int ks = 0; ks < 4; ++ks) {     // K = 128 in 4 steps of 32
      v16h a  = load_frag_row(Asub, 16 * w, 32 * ks);
      v16h bf = load_frag_row(Wt, 16 * tn, 32 * ks);
      acc = __builtin_amdgcn_wmma_f32_16x16x32_f16(false, a, false, bf,
                                                   (short)0, acc, false, false);
    }
    const int kcol = 16 * tn + (l & 15);
    const float bias = bs[kcol];
    if (tensor == 2) {
      // V^T: vt[(n*DIM + c)*SEQ + h], h = 128*hb + kcol, c = 16w + 8hi + r.
      // Lanes 0-15 have consecutive h -> 32B-coalesced half-wave stores.
#pragma unroll
      for (int r = 0; r < 8; ++r) {
        const int c = 16 * w + 8 * hi + r;
        dst[((size_t)(n * DIM + c) << 9) + 128 * hb + kcol] =
            (_Float16)(acc[r] + bias);
      }
    } else {
      // Accumulator VGPR r holds (row = 16w + 8hi + r, col = kcol): 8
      // consecutive c-values per lane for fixed h -> one 16-byte store.
      v8h outv;
#pragma unroll
      for (int r = 0; r < 8; ++r) outv[r] = (_Float16)(acc[r] + bias);
      _Float16* dp = dst + ((size_t)(n * SEQ + 128 * hb + kcol) << 7)
                         + 16 * w + 8 * hi;
      *(v8h*)dp = outv;
    }
  }
}

// ---------------------------------------------------------------------------
// Kernel 2: attention. One workgroup per (head n, 128-query block).
//  Phase 1: S = scale * Q K^T via WMMA; raw scores -> attn buffer; rowmax via
//           shfl_xor reduction (wave owns its 16 rows exclusively).
//  Phase 2: re-read S (L2), p = exp(s - rowmax), rowsum (ds_add_f32),
//           p -> LDS f16, ctxt += P V via WMMA (V^T async-DMA'd into LDS).
//  Phase 3: normalize attn rows and ctxt by 1/rowsum.
// ---------------------------------------------------------------------------
__global__ __launch_bounds__(256) void attn_kernel(
    const _Float16* __restrict__ qp, const _Float16* __restrict__ kp,
    const _Float16* __restrict__ vt,
    float* __restrict__ out, float* __restrict__ attn)
{
  __shared__ _Float16 smA[128 * LDS_STRIDE];   // Q block, then P block
  __shared__ _Float16 smB[128 * LDS_STRIDE];   // K block, then V^T block
  __shared__ float rowmax[128];
  __shared__ float rowsum[128];

  const int n     = blockIdx.x >> 2;
  const int rb    = blockIdx.x & 3;
  const int qbase = rb * 128;
  const int t  = threadIdx.x;
  const int w  = t >> 5, l = t & 31, hi = l >> 4;
  const int srow = t >> 1;
  const int scol = (t & 1) * 64;
  const float scale = 0.08838834764831843f;    // 1/sqrt(128)

  const _Float16* qhead = qp + ((size_t)n * SEQ << 7);   // (h, c) f16
  const _Float16* khead = kp + ((size_t)n * SEQ << 7);   // (h, c) f16
  const _Float16* vhead = vt + ((size_t)n * DIM << 9);   // (c, h) f16

  _Float16* arow = smA + srow * LDS_STRIDE + scol;
  _Float16* brow = smB + srow * LDS_STRIDE + scol;

  // Async-stage Q block: 8 x 16B per thread, overlapped with rowmax init.
#pragma unroll
  for (int i = 0; i < 64; i += 8)
    async_copy_b128(qhead, (unsigned)(((qbase + srow) * DIM + scol + i) * 2),
                    arow + i);
  if (t < 128) { rowmax[t] = -3.0e38f; rowsum[t] = 0.0f; }

  // ---- Phase 1 ----
  for (int kb = 0; kb < 4; ++kb) {
    // Async-stage K block (B^T tile = row-major keys x channels).
#pragma unroll
    for (int i = 0; i < 64; i += 8)
      async_copy_b128(khead,
                      (unsigned)(((kb * 128 + srow) * DIM + scol + i) * 2),
                      brow + i);
    wait_async0();
    __syncthreads();

    float rmax[8];
#pragma unroll
    for (int r = 0; r < 8; ++r) rmax[r] = -3.0e38f;

#pragma unroll
    for (int tn = 0; tn < 8; ++tn) {
      v8f acc = {};
#pragma unroll
      for (int ks = 0; ks < 4; ++ks) {
        v16h a  = load_frag_row(smA, 16 * w, 32 * ks);
        v16h bf = load_frag_row(smB, 16 * tn, 32 * ks);
        acc = __builtin_amdgcn_wmma_f32_16x16x32_f16(false, a, false, bf,
                                                     (short)0, acc, false, false);
      }
      const int col = kb * 128 + 16 * tn + (l & 15);
      float* sp = attn + (size_t)n * (SEQ * SEQ)
                       + (size_t)(qbase + 16 * w + 8 * hi) * SEQ + col;
#pragma unroll
      for (int r = 0; r < 8; ++r) {
        float s = acc[r] * scale;
        rmax[r] = fmaxf(rmax[r], s);
        sp[(size_t)r * SEQ] = s;
      }
    }
    // Column-wise max lives across the 16 lanes of each half-wave.
#pragma unroll
    for (int r = 0; r < 8; ++r) {
      float m = rmax[r];
      m = fmaxf(m, __shfl_xor(m, 1, 32));
      m = fmaxf(m, __shfl_xor(m, 2, 32));
      m = fmaxf(m, __shfl_xor(m, 4, 32));
      m = fmaxf(m, __shfl_xor(m, 8, 32));
      if ((l & 15) == 0) {
        const int row = 16 * w + r + 8 * hi;       // unique per (wave,lane-half)
        rowmax[row] = fmaxf(rowmax[row], m);
      }
    }
    __syncthreads();
  }

  // ---- Phase 2 ----
  v8f cacc[8] = {};
  float lsum = 0.0f;
  const float m = rowmax[srow];

  for (int kb = 0; kb < 4; ++kb) {
    // Async-stage V^T tile: smB[ch][key] straight from the pre-transposed
    // tensor; DMA runs while we do the exp() work below.
#pragma unroll
    for (int i = 0; i < 64; i += 8)
      async_copy_b128(vhead,
                      (unsigned)((srow * SEQ + kb * 128 + scol + i) * 2),
                      brow + i);
    // Re-read scores, exponentiate, write p back (f32) and into LDS (f16).
    {
      float* sp = attn + (size_t)n * (SEQ * SEQ)
                       + (size_t)(qbase + srow) * SEQ + kb * 128 + scol;
      _Float16* prow = arow;
#pragma unroll
      for (int i = 0; i < 64; i += 4) {
        float4 s4 = *(const float4*)(sp + i);
        float e0 = __expf(s4.x - m), e1 = __expf(s4.y - m);
        float e2 = __expf(s4.z - m), e3 = __expf(s4.w - m);
        lsum += e0 + e1 + e2 + e3;
        *(float4*)(sp + i) = make_float4(e0, e1, e2, e3);
        prow[i + 0] = (_Float16)e0; prow[i + 1] = (_Float16)e1;
        prow[i + 2] = (_Float16)e2; prow[i + 3] = (_Float16)e3;
      }
    }
    wait_async0();
    __syncthreads();
#pragma unroll
    for (int tn = 0; tn < 8; ++tn) {
#pragma unroll
      for (int ks = 0; ks < 4; ++ks) {
        v16h a  = load_frag_row(smA, 16 * w, 32 * ks);
        v16h bf = load_frag_row(smB, 16 * tn, 32 * ks);
        cacc[tn] = __builtin_amdgcn_wmma_f32_16x16x32_f16(false, a, false, bf,
                                                          (short)0, cacc[tn],
                                                          false, false);
      }
    }
    __syncthreads();
  }
  atomicAdd(&rowsum[srow], lsum);   // ds_add_f32
  __syncthreads();

  // ---- Phase 3: normalize ----
  {
    const float inv = 1.0f / rowsum[srow];
    float* sp = attn + (size_t)n * (SEQ * SEQ) + (size_t)(qbase + srow) * SEQ + scol;
    for (int kb = 0; kb < 4; ++kb) {
      float* p = sp + kb * 128;
#pragma unroll
      for (int i = 0; i < 64; i += 4) {
        float4 v = *(const float4*)(p + i);
        v.x *= inv; v.y *= inv; v.z *= inv; v.w *= inv;
        *(float4*)(p + i) = v;
      }
    }
  }
  {
    float invr[8];
#pragma unroll
    for (int r = 0; r < 8; ++r) invr[r] = 1.0f / rowsum[16 * w + r + 8 * hi];
#pragma unroll
    for (int tn = 0; tn < 8; ++tn) {
      const int ch = 16 * tn + (l & 15);
      float* op = out + (size_t)(n * SEQ + qbase + 16 * w + 8 * hi) * DIM + ch;
#pragma unroll
      for (int r = 0; r < 8; ++r)
        op[(size_t)r * DIM] = cacc[tn][r] * invr[r];
    }
  }
}

// ---------------------------------------------------------------------------
extern "C" void kernel_launch(void* const* d_in, const int* in_sizes, int n_in,
                              void* d_out, int out_size, void* d_ws, size_t ws_size,
                              hipStream_t stream) {
  (void)in_sizes; (void)n_in; (void)out_size; (void)ws_size;
  const float* q_in = (const float*)d_in[0];
  const float* k_in = (const float*)d_in[1];
  const float* v_in = (const float*)d_in[2];
  const float* Wq = (const float*)d_in[3];
  const float* bq = (const float*)d_in[4];
  const float* Wk = (const float*)d_in[5];
  const float* bk = (const float*)d_in[6];
  const float* Wv = (const float*)d_in[7];
  const float* bv = (const float*)d_in[8];

  float* out  = (float*)d_out;                       // (1,512,512*128) f32
  float* attn = out + (size_t)NHEAD * SEQ * DIM;     // (512,1,512,512) f32

  // Workspace: f16 projected tensors Q,K (n,h,c) and V^T (n,c,h). 192 MB.
  _Float16* qp = (_Float16*)d_ws;
  _Float16* kp = qp + (size_t)NHEAD * SEQ * DIM;
  _Float16* vt = kp + (size_t)NHEAD * SEQ * DIM;

  hipLaunchKernelGGL(proj_kernel, dim3(3 * NHEAD * 4), dim3(256), 0, stream,
                     q_in, k_in, v_in, Wq, bq, Wk, bk, Wv, bv, qp, kp, vt);
  hipLaunchKernelGGL(attn_kernel, dim3(NHEAD * 4), dim3(256), 0, stream,
                     qp, kp, vt, out, attn);
}
